// VectorQuantizer_60524679135528
// MI455X (gfx1250) — matrix operand order repaired and checked by reading, not compiled, not run on hardware
//
#include <hip/hip_runtime.h>

typedef __attribute__((ext_vector_type(16))) _Float16 v16h;
typedef __attribute__((ext_vector_type(8)))  _Float16 v8h;
typedef __attribute__((ext_vector_type(8)))  float    v8f;

#define EMB_DIM    256
#define NUM_EMB    1024
#define POS_TILE   128
#define CODE_CHUNK 64
#define N_CHUNKS   16
#define LDS_ROW    264         // halfs per row: 256 + 8 pad (16B) vs bank conflicts
#define ROW_BYTES  528
#define HW         4096        // 64*64
#define ZQ_ELEMS   33554432    // 32*256*64*64

// LDS layout (bytes)
#define OFF_ZHI    0u
#define OFF_ZLO    67584u
#define OFF_EBUF   135168u     // 2 buffers x (hi 33792 + lo 33792)
#define EBUF_SZ    67584u
#define ELO_SKIP   33792u
#define OFF_ENORM  270336u     // f32[1024]
#define OFF_ZNORM  274432u     // f32[128]
#define OFF_DIST   274944u     // f32[128]
#define OFF_NPART  275456u     // f32[256]
#define OFF_IDX    276480u     // i32[128]
#define LDS_TOTAL  276992u

union V16 { v16h v; v8h h[2]; };

// Async global->LDS copy, 16B per lane, GVS addressing (CDNA5, ASYNCcnt).
static __device__ __forceinline__ void async_cp_b128(unsigned lds_off,
                                                     const void* sbase,
                                                     unsigned voff) {
  asm volatile("global_load_async_to_lds_b128 %0, %1, %2"
               :: "v"(lds_off), "v"(voff), "s"(sbase)
               : "memory");
}

// ---------------------------------------------------------------------------
// Kernel 1: codebook prep.  emb (f32 [1024][256]) -> f16 hi/lo split + ||e||^2
// ---------------------------------------------------------------------------
__global__ __launch_bounds__(256) void vq_prep(const float* __restrict__ emb,
                                               _Float16* __restrict__ ehi,
                                               _Float16* __restrict__ elo,
                                               float* __restrict__ enorm) {
  const int k = blockIdx.x;
  const int t = threadIdx.x;
  float v = emb[k * EMB_DIM + t];
  _Float16 h = (_Float16)v;
  _Float16 l = (_Float16)(v - (float)h);
  ehi[k * EMB_DIM + t] = h;
  elo[k * EMB_DIM + t] = l;
  float s = v * v;
  #pragma unroll
  for (int off = 16; off; off >>= 1) s += __shfl_xor(s, off, 32);
  __shared__ float red[8];
  if ((t & 31) == 0) red[t >> 5] = s;
  __syncthreads();
  if (t == 0) {
    float tot = 0.f;
    #pragma unroll
    for (int i = 0; i < 8; ++i) tot += red[i];
    enorm[k] = tot;
  }
}

// ---------------------------------------------------------------------------
// Kernel 2: main VQ.  128 positions/WG vs 1024 codes via f16 WMMA (fp16x3),
// codebook chunks double-buffered with global_load_async_to_lds_b128.
// ---------------------------------------------------------------------------
__global__ __launch_bounds__(256) void vq_main(const float* __restrict__ z,
                                               const float* __restrict__ emb,
                                               const _Float16* __restrict__ ehi_g,
                                               const _Float16* __restrict__ elo_g,
                                               const float* __restrict__ enorm_g,
                                               float* __restrict__ out_zq,
                                               float* __restrict__ partials) {
  extern __shared__ char smem[];
  _Float16* zhi     = (_Float16*)(smem + OFF_ZHI);     // [128][264]
  _Float16* zlo     = (_Float16*)(smem + OFF_ZLO);     // [128][264]
  float*    enorm_l = (float*)   (smem + OFF_ENORM);   // [1024]
  float*    znorm_l = (float*)   (smem + OFF_ZNORM);   // [128]
  float*    dist_l  = (float*)   (smem + OFF_DIST);    // [128]
  float*    npart   = (float*)   (smem + OFF_NPART);   // [256]
  int*      idx_l   = (int*)     (smem + OFF_IDX);     // [128]
  const unsigned lds_base = (unsigned)(size_t)(void*)smem;

  const int t   = threadIdx.x;
  const int wg  = blockIdx.x;          // 1024 WGs
  const int b   = wg >> 5;             // batch image
  const int hw0 = (wg & 31) * POS_TILE;

  // ---- prologue: issue async copies for chunk 0 (both hi and lo halves)
  {
    const unsigned lofs = (unsigned)((t >> 5) * ROW_BYTES + (t & 31) * 16);
    const unsigned vofs = (unsigned)(t * 16);
    #pragma unroll
    for (int i = 0; i < 8; ++i) {
      async_cp_b128(lds_base + OFF_EBUF + lofs + i * 4224u, ehi_g, vofs + i * 4096u);
      async_cp_b128(lds_base + OFF_EBUF + ELO_SKIP + lofs + i * 4224u, elo_g, vofs + i * 4096u);
    }
  }

  // ---- stage all 1024 code norms (4 KB) once
  *(float4*)&enorm_l[t * 4] = ((const float4*)enorm_g)[t];

  // ---- load z tile (coalesced along spatial), f16 hi/lo into LDS, ||z||^2
  {
    const int p = t & 127;
    const int hsel = t >> 7;
    const float* zbase = z + (size_t)b * EMB_DIM * HW + hw0 + p;
    float nacc = 0.f;
    #pragma unroll 4
    for (int i = 0; i < 128; ++i) {
      const int d = 2 * i + hsel;
      float v = zbase[(size_t)d * HW];
      _Float16 h = (_Float16)v;
      _Float16 l = (_Float16)(v - (float)h);
      zhi[p * LDS_ROW + d] = h;
      zlo[p * LDS_ROW + d] = l;
      nacc += v * v;
    }
    npart[t] = nacc;
  }
  __syncthreads();
  if (t < 128) znorm_l[t] = npart[t] + npart[t + 128];

  const int lane  = t & 31;
  const int wv    = t >> 5;
  const int m     = lane & 15;   // A row (code) / B column (position)
  const int g     = lane >> 4;   // half-wave group
  const int nbase = wv * 16;     // this wave's 16 positions

  const unsigned lofs = (unsigned)((t >> 5) * ROW_BYTES + (t & 31) * 16);
  const unsigned vofs = (unsigned)(t * 16);

  float best = 3.4e38f;
  int   bidx = 0;

  for (int c = 0; c < N_CHUNKS; ++c) {
    // ---- issue chunk c+1 into the other buffer, then wait for chunk c.
    // Each wave issued exactly 16 asyncs per chunk; completion is in-order,
    // so ASYNCcnt<=16 guarantees chunk c has fully landed in LDS.
    if (c < N_CHUNKS - 1) {
      const unsigned nb = OFF_EBUF + (unsigned)((c + 1) & 1) * EBUF_SZ;
      const _Float16* shi = ehi_g + (size_t)(c + 1) * CODE_CHUNK * EMB_DIM;
      const _Float16* slo = elo_g + (size_t)(c + 1) * CODE_CHUNK * EMB_DIM;
      #pragma unroll
      for (int i = 0; i < 8; ++i) {
        async_cp_b128(lds_base + nb + lofs + i * 4224u, shi, vofs + i * 4096u);
        async_cp_b128(lds_base + nb + ELO_SKIP + lofs + i * 4224u, slo, vofs + i * 4096u);
      }
      asm volatile("s_wait_asynccnt 0x10" ::: "memory");
    } else {
      asm volatile("s_wait_asynccnt 0x0" ::: "memory");
    }
    __syncthreads();   // chunk c visible to every wave

    const _Float16* Ehi = (const _Float16*)(smem + OFF_EBUF + (unsigned)(c & 1) * EBUF_SZ);
    const _Float16* Elo = (const _Float16*)((const char*)Ehi + ELO_SKIP);

    v8f acc[4] = {};   // 4 independent code-tile chains

    #pragma unroll
    for (int ks = 0; ks < 8; ++ks) {
      const int kb = ks * 32;
      V16 bhi, blo;
      {
        const _Float16* zp = &zhi[(nbase + m) * LDS_ROW + kb + g * 16];
        bhi.h[0] = *(const v8h*)zp;  bhi.h[1] = *(const v8h*)(zp + 8);
        const _Float16* zq = &zlo[(nbase + m) * LDS_ROW + kb + g * 16];
        blo.h[0] = *(const v8h*)zq;  blo.h[1] = *(const v8h*)(zq + 8);
      }
      #pragma unroll
      for (int ct = 0; ct < 4; ++ct) {
        const _Float16* ep = &Ehi[(ct * 16 + m) * LDS_ROW + kb + g * 16];
        V16 ahi; ahi.h[0] = *(const v8h*)ep;  ahi.h[1] = *(const v8h*)(ep + 8);
        const _Float16* eq = &Elo[(ct * 16 + m) * LDS_ROW + kb + g * 16];
        V16 alo; alo.h[0] = *(const v8h*)eq;  alo.h[1] = *(const v8h*)(eq + 8);
        // S += Zhi*Ehi + Zhi*Elo + Zlo*Ehi   (fp16x3, ~f32-accurate)
        acc[ct] = __builtin_amdgcn_wmma_f32_16x16x32_f16(false, ahi.v, false, bhi.v,
                                                         (short)0, acc[ct], false, false);
        acc[ct] = __builtin_amdgcn_wmma_f32_16x16x32_f16(false, ahi.v, false, blo.v,
                                                         (short)0, acc[ct], false, false);
        acc[ct] = __builtin_amdgcn_wmma_f32_16x16x32_f16(false, alo.v, false, bhi.v,
                                                         (short)0, acc[ct], false, false);
      }
    }

    // ---- score = ||e||^2 - 2*S ; lane-local running argmin (codes ascend)
    #pragma unroll
    for (int ct = 0; ct < 4; ++ct) {
      const float* enp = &enorm_l[c * CODE_CHUNK + ct * 16 + g * 8];
      const float4 ena = *(const float4*)enp;
      const float4 enb = *(const float4*)(enp + 4);
      const float en[8] = {ena.x, ena.y, ena.z, ena.w, enb.x, enb.y, enb.z, enb.w};
      #pragma unroll
      for (int v = 0; v < 8; ++v) {
        const float score = en[v] - 2.0f * acc[ct][v];
        const int code = c * CODE_CHUNK + ct * 16 + g * 8 + v;
        if (score < best) { best = score; bidx = code; }
      }
    }
    __syncthreads();   // all waves done reading buf (c&1) before it is re-filled
  }

  // merge the two half-wave halves; tie-break on lower index (jnp.argmin)
  {
    const float ob = __shfl_xor(best, 16, 32);
    const int   oi = __shfl_xor(bidx, 16, 32);
    if (ob < best || (ob == best && oi < bidx)) { best = ob; bidx = oi; }
  }
  if (lane < 16) { idx_l[nbase + lane] = bidx; dist_l[nbase + lane] = best; }
  __syncthreads();

  // deterministic per-WG loss partial: sum_p (min_score_p + ||z_p||^2)
  if (t == 0) {
    float s = 0.f;
    for (int i = 0; i < 128; ++i) s += dist_l[i] + znorm_l[i];
    partials[wg] = s;
  }

  // ---- gather output z_q = emb[idx] back to NCHW (coalesced stores)
  {
    const int p = t & 127;
    const int hsel = t >> 7;
    const int code = idx_l[p];
    const float* erow = emb + (size_t)code * EMB_DIM;   // L2-resident gather
    float* obase = out_zq + (size_t)b * EMB_DIM * HW + hw0 + p;
    #pragma unroll 4
    for (int i = 0; i < 128; ++i) {
      const int cch = 2 * i + hsel;
      obase[(size_t)cch * HW] = erow[cch];
    }
  }
}

// ---------------------------------------------------------------------------
// Kernel 3: fixed-tree loss reduction (deterministic)
// loss = (BETA + 1) * mean((z_q - z)^2) = 1.25 * total / (N*D)
// ---------------------------------------------------------------------------
__global__ __launch_bounds__(256) void vq_loss(const float* __restrict__ partials,
                                               float* __restrict__ out_loss) {
  __shared__ float red[256];
  const int t = threadIdx.x;
  red[t] = partials[t] + partials[t + 256] + partials[t + 512] + partials[t + 768];
  __syncthreads();
  #pragma unroll
  for (int off = 128; off; off >>= 1) {
    if (t < off) red[t] += red[t + off];
    __syncthreads();
  }
  if (t == 0) out_loss[0] = 1.25f * red[0] / 33554432.0f;
}

// ---------------------------------------------------------------------------
extern "C" void kernel_launch(void* const* d_in, const int* in_sizes, int n_in,
                              void* d_out, int out_size, void* d_ws, size_t ws_size,
                              hipStream_t stream) {
  (void)in_sizes; (void)n_in; (void)out_size; (void)ws_size;
  const float* z   = (const float*)d_in[0];
  const float* emb = (const float*)d_in[1];
  float* out = (float*)d_out;

  char* ws = (char*)d_ws;
  _Float16* ehi   = (_Float16*)(ws);                      // 512 KB
  _Float16* elo   = (_Float16*)(ws + 512 * 1024);         // 512 KB
  float*    enorm = (float*)   (ws + 1024 * 1024);        // 4 KB
  float*    parts = (float*)   (ws + 1024 * 1024 + 4096); // 4 KB

  vq_prep<<<NUM_EMB, 256, 0, stream>>>(emb, ehi, elo, enorm);

  const size_t lds_bytes = LDS_TOTAL;   // ~270 KB of the WGP's 320 KB
  hipFuncSetAttribute((const void*)vq_main,
                      hipFuncAttributeMaxDynamicSharedMemorySize, (int)lds_bytes);
  vq_main<<<1024, 256, lds_bytes, stream>>>(z, emb, ehi, elo, enorm, out, parts);

  vq_loss<<<1, 256, 0, stream>>>(parts, out + ZQ_ELEMS);
}
